// MultiHeadAttention_90769838834143
// MI455X (gfx1250) — compile-verified, matrix operand order checked by reading, not run on hardware
//
#include <hip/hip_runtime.h>
#include <hip/hip_bf16.h>

// ---------------------------------------------------------------------------
// Problem constants: B=1, S=4096, D=768, H=12, DH=64
// ---------------------------------------------------------------------------
constexpr int S    = 4096;
constexpr int D    = 768;
constexpr int H    = 12;
constexpr int DH   = 64;
constexpr int KDIM = D;                 // GEMM K
constexpr int NDIM = D;                 // GEMM N
constexpr int KSTEP = 32;               // WMMA K per step (bf16)
constexpr int NKSTEPS = KDIM / KSTEP;   // 24

typedef __attribute__((ext_vector_type(16))) __bf16 v16bf;
typedef __attribute__((ext_vector_type(8)))  __bf16 v8bf;
typedef __attribute__((ext_vector_type(8)))  float  v8f;

union Frag16 { v16bf v; v8bf h[2]; };

__device__ __forceinline__ v8f wmma_bf16(v16bf a, v16bf b, v8f c) {
    return __builtin_amdgcn_wmma_f32_16x16x32_bf16(
        false, a, false, b, (short)0, c, false, false);
}

__device__ __forceinline__ float rowmax16(float v) {
    v = fmaxf(v, __shfl_xor(v, 1, 32));
    v = fmaxf(v, __shfl_xor(v, 2, 32));
    v = fmaxf(v, __shfl_xor(v, 4, 32));
    v = fmaxf(v, __shfl_xor(v, 8, 32));
    return v;
}
__device__ __forceinline__ float rowsum16(float v) {
    v += __shfl_xor(v, 1, 32);
    v += __shfl_xor(v, 2, 32);
    v += __shfl_xor(v, 4, 32);
    v += __shfl_xor(v, 8, 32);
    return v;
}

// ---- CDNA5 async global->LDS copy (ASYNCcnt path), GVS addressing ---------
__device__ __forceinline__ void async_copy16(uint32_t lds_off_bytes,
                                             const void* sbase,
                                             uint32_t goff_bytes) {
    asm volatile("global_load_async_to_lds_b128 %0, %1, %2"
                 :: "v"(lds_off_bytes), "v"(goff_bytes),
                    "s"((uint64_t)(uintptr_t)sbase)
                 : "memory");
}
// wait until all outstanding async ops done
__device__ __forceinline__ void wait_async0() {
    asm volatile("s_wait_asynccnt 0x0" ::: "memory");
}
// wait until only the newest group of 4 async ops remains in flight
__device__ __forceinline__ void wait_async4() {
    asm volatile("s_wait_asynccnt 0x4" ::: "memory");
}
__device__ __forceinline__ uint32_t lds_u32(const void* p) {
    return (uint32_t)(uintptr_t)p;   // low 32 bits of generic LDS addr = LDS offset
}

// ---------------------------------------------------------------------------
// Kernel 1: pack fp32 weights (K x N row-major) into bf16 WMMA B-operand
// fragments: frag(ntile,ktile) = 512 bf16; lane l owns 16 contiguous bf16.
//   lane <16 : k in {0..7, 16..23};  lane >=16 : k in {8..15, 24..31}
// ---------------------------------------------------------------------------
__global__ void pack_weights(const float* __restrict__ W0, const float* __restrict__ W1,
                             const float* __restrict__ W2, const float* __restrict__ W3,
                             __bf16* __restrict__ out) {
    int id = blockIdx.x * blockDim.x + threadIdx.x;
    const int total = 4 * (NDIM / 16) * NKSTEPS * 32;
    if (id >= total) return;
    int lane = id & 31;
    int kt   = (id >> 5) % NKSTEPS;
    int ntf  = ((id >> 5) / NKSTEPS) % (NDIM / 16);
    int w    = (id >> 5) / (NKSTEPS * (NDIM / 16));
    const float* W = (w == 0) ? W0 : (w == 1) ? W1 : (w == 2) ? W2 : W3;
    int half = lane >> 4, lr = lane & 15;
    int n = ntf * 16 + lr;
    __bf16* dst = out + (size_t)w * KDIM * NDIM + ((size_t)ntf * NKSTEPS + kt) * 512 + lane * 16;
#pragma unroll
    for (int j = 0; j < 16; ++j) {
        int k = kt * KSTEP + half * 8 + ((j < 8) ? j : (8 + j));
        dst[j] = (__bf16)W[(size_t)k * NDIM + n];
    }
}

// ---------------------------------------------------------------------------
// Kernel 1b: fp32 -> bf16 bulk conversion (for x)
// ---------------------------------------------------------------------------
__global__ void convert_bf16(const float* __restrict__ in, __bf16* __restrict__ out, int n4) {
    int id = blockIdx.x * blockDim.x + threadIdx.x;
    if (id >= n4) return;
    float4 f = ((const float4*)in)[id];
    __bf16* d = out + (size_t)id * 4;
    d[0] = (__bf16)f.x; d[1] = (__bf16)f.y; d[2] = (__bf16)f.z; d[3] = (__bf16)f.w;
}

// ---------------------------------------------------------------------------
// Kernel 2: GEMM  C[M x 768](fp32) = A[M x 768](bf16) * Bpack(bf16 frags)
// 4 waves (2x2) -> 128x128 block tile; wave tile 64x64 (16 accumulators).
// A tiles double-buffered in LDS, filled by async global->LDS b128 copies
// that overlap the WMMA compute of the previous k-step.
// ---------------------------------------------------------------------------
constexpr int ALDS_STRIDE = 40;   // bf16 elems; 80B rows: 16B-aligned, bank-spread

__global__ __launch_bounds__(128)
void gemm_bf16(const __bf16* __restrict__ A, const __bf16* __restrict__ Bp,
               float* __restrict__ C, int M) {
    __shared__ __bf16 Alds[2][128 * ALDS_STRIDE];
    const int tid  = threadIdx.x;
    const int lane = tid & 31;
    const int wave = tid >> 5;
    const int wr   = wave >> 1;
    const int wc   = wave & 1;
    const int half = lane >> 4, lr = lane & 15;
    const int mblk = blockIdx.x * 128;
    const int nfragBase = blockIdx.y * 8 + wc * 4;
    const uint32_t aldsbase = lds_u32(&Alds[0][0]);
    const uint32_t abufbytes = 128 * ALDS_STRIDE * 2;

    auto stageA = [&](int kt, int buf) {
#pragma unroll
        for (int c = 0; c < 4; ++c) {
            async_copy16(aldsbase + buf * abufbytes + (uint32_t)(tid * ALDS_STRIDE + c * 8) * 2,
                         A,
                         (uint32_t)(((mblk + tid) * KDIM + kt * KSTEP + c * 8) * 2));
        }
    };

    v8f acc[4][4] = {};

    stageA(0, 0);
    for (int kt = 0; kt < NKSTEPS; ++kt) {
        const int buf = kt & 1;
        if (kt + 1 < NKSTEPS) {           // prefetch next tile into other buffer
            stageA(kt + 1, buf ^ 1);
            wait_async4();                // oldest group (this tile) has landed
        } else {
            wait_async0();
        }
        __syncthreads();

        // ---- A fragments (4 m-tiles)
        v16bf afr[4];
#pragma unroll
        for (int mt = 0; mt < 4; ++mt) {
            const __bf16* rp = &Alds[buf][0] + (wr * 64 + mt * 16 + lr) * ALDS_STRIDE + half * 8;
            Frag16 f;
            f.h[0] = *(const v8bf*)(rp);
            f.h[1] = *(const v8bf*)(rp + 16);
            afr[mt] = f.v;
        }

        // ---- B fragments straight from global packed buffer, 16 WMMAs
#pragma unroll
        for (int nt = 0; nt < 4; ++nt) {
            const __bf16* bp = Bp + ((size_t)(nfragBase + nt) * NKSTEPS + kt) * 512 + lane * 16;
            Frag16 b;
            b.h[0] = *(const v8bf*)(bp);
            b.h[1] = *(const v8bf*)(bp + 8);
#pragma unroll
            for (int mt = 0; mt < 4; ++mt)
                acc[mt][nt] = wmma_bf16(afr[mt], b.v, acc[mt][nt]);
        }
        __syncthreads();                  // reads done before buffer reuse at kt+2
    }

#pragma unroll
    for (int mt = 0; mt < 4; ++mt)
#pragma unroll
        for (int nt = 0; nt < 4; ++nt)
#pragma unroll
            for (int r = 0; r < 8; ++r) {
                int row = mblk + wr * 64 + mt * 16 + r + half * 8;
                int col = blockIdx.y * 128 + wc * 64 + nt * 16 + lr;
                C[(size_t)row * NDIM + col] = acc[mt][nt][r];
            }
}

// ---------------------------------------------------------------------------
// Kernel 3: RoPE on Q,K (-> bf16 [S][D]) + V -> bf16 transposed [H*DH][S]
// ---------------------------------------------------------------------------
__global__ void rope_pack(const float* __restrict__ qr, const float* __restrict__ kr,
                          const float* __restrict__ vr, const float* __restrict__ freqs,
                          __bf16* __restrict__ Qb, __bf16* __restrict__ Kb,
                          __bf16* __restrict__ Vt) {
    int id = blockIdx.x * blockDim.x + threadIdx.x;
    if (id >= S * H * (DH / 2)) return;
    int i = id & 31;
    int h = (id >> 5) % H;
    int s = id / (H * (DH / 2));
    float f  = freqs[s * (DH / 2) + i];
    float cf = cosf(f), sf = sinf(f);
    size_t base = (size_t)s * D + h * DH + 2 * i;
    float q0 = qr[base], q1 = qr[base + 1];
    Qb[base]     = (__bf16)(q0 * cf - q1 * sf);
    Qb[base + 1] = (__bf16)(q0 * sf + q1 * cf);
    float k0 = kr[base], k1 = kr[base + 1];
    Kb[base]     = (__bf16)(k0 * cf - k1 * sf);
    Kb[base + 1] = (__bf16)(k0 * sf + k1 * cf);
    Vt[(size_t)(h * DH + 2 * i) * S + s]     = (__bf16)vr[base];
    Vt[(size_t)(h * DH + 2 * i + 1) * S + s] = (__bf16)vr[base + 1];
}

// ---------------------------------------------------------------------------
// Kernel 4: causal flash attention, 64-key blocks, double-buffered K/V tiles.
// Grid (32 qblocks, 12 heads), 256 thr = 8 waves; wave owns 16 query rows.
// K tile [64 key][64 dh] and V tile [64 dh][64 key] staged via async copies
// that overlap the previous block's WMMA + softmax work.
// Per key block: 8 score WMMAs + 8 PV WMMAs.  Output written as bf16 [S][D].
// ---------------------------------------------------------------------------
constexpr int KSTR = 72;   // 144B rows: 16B-aligned, bank-spread
constexpr int VSTR = 72;
constexpr int PSTR = 72;

__global__ __launch_bounds__(256)
void attention(const __bf16* __restrict__ Qb, const __bf16* __restrict__ Kb,
               const __bf16* __restrict__ Vt, __bf16* __restrict__ aout) {
    __shared__ __bf16 Klds[2][64 * KSTR];   // [key][dh]
    __shared__ __bf16 Vlds[2][64 * VSTR];   // [dh][key]
    __shared__ __bf16 Plds[8 * 16 * PSTR];  // per-wave P tile [m][key]

    const int tid = threadIdx.x, lane = tid & 31, wave = tid >> 5;
    const int half = lane >> 4, lr = lane & 15;
    const int head = blockIdx.y;
    const int qb = blockIdx.x;
    const int qbase = qb * 128 + wave * 16;
    const int colbase = head * DH;
    const float scale = 0.125f;   // DH^-0.5
    const uint32_t kldsbase = lds_u32(&Klds[0][0]);
    const uint32_t vldsbase = lds_u32(&Vlds[0][0]);
    const uint32_t kbufbytes = 64 * KSTR * 2;
    const uint32_t vbufbytes = 64 * VSTR * 2;

    // staging: K (8KB) + Vt (8KB); 512 16B-units each, 2 of each per thread
    auto stageKV = [&](int kbase, int buf) {
#pragma unroll
        for (int i = 0; i < 2; ++i) {
            int u = tid + i * 256;
            int row = u >> 3, c8 = (u & 7) * 8;
            async_copy16(kldsbase + buf * kbufbytes + (uint32_t)(row * KSTR + c8) * 2, Kb,
                         (uint32_t)(((kbase + row) * D + colbase + c8) * 2));
            async_copy16(vldsbase + buf * vbufbytes + (uint32_t)(row * VSTR + c8) * 2, Vt,
                         (uint32_t)(((colbase + row) * S + kbase + c8) * 2));
        }
    };

    // Q operand fragments (16 x 64), loaded once
    v16bf qf[2];
#pragma unroll
    for (int ks = 0; ks < 2; ++ks) {
        const __bf16* p = Qb + (size_t)(qbase + lr) * D + colbase + ks * 32 + half * 8;
        Frag16 f;
        f.h[0] = *(const v8bf*)(p);
        f.h[1] = *(const v8bf*)(p + 16);
        qf[ks] = f.v;
    }

    v8f o[4] = {};
    float mi[8], li[8];
#pragma unroll
    for (int r = 0; r < 8; ++r) { mi[r] = -1e30f; li[r] = 0.0f; }

    const int nkb = qb * 2 + 2;          // causal: 64-key blocks
    const int q_hi = qbase + 15;
    __bf16* Pw = Plds + wave * 16 * PSTR;

    stageKV(0, 0);
    for (int kb = 0; kb < nkb; ++kb) {
        const int kbase = kb * 64;
        const int buf = kb & 1;
        if (kb + 1 < nkb) {               // prefetch next K/V tiles
            stageKV(kbase + 64, buf ^ 1);
            wait_async4();                // this block's tiles have landed
        } else {
            wait_async0();
        }
        __syncthreads();

        if (kbase <= q_hi) {              // wave-uniform causal skip
            // ---- scores S = Q K^T  (16 x 64): 8 WMMAs
            v8f sc[4] = {};
#pragma unroll
            for (int ks = 0; ks < 2; ++ks)
#pragma unroll
                for (int nt = 0; nt < 4; ++nt) {
                    const __bf16* kp = &Klds[buf][0] + (nt * 16 + lr) * KSTR + ks * 32 + half * 8;
                    Frag16 b;
                    b.h[0] = *(const v8bf*)(kp);
                    b.h[1] = *(const v8bf*)(kp + 16);
                    sc[nt] = wmma_bf16(qf[ks], b.v, sc[nt]);
                }

            // ---- online softmax (rows at vgpr r within half-wave)
            float pv[4][8], csc[8];
#pragma unroll
            for (int r = 0; r < 8; ++r) {
                int q = qbase + r + half * 8;
                float vals[4];
#pragma unroll
                for (int nt = 0; nt < 4; ++nt) {
                    float v = sc[nt][r] * scale;
                    if (kbase + nt * 16 + lr > q) v = -1e30f;
                    vals[nt] = v;
                }
                float mx = fmaxf(fmaxf(vals[0], vals[1]), fmaxf(vals[2], vals[3]));
                mx = rowmax16(mx);
                float mn = fmaxf(mi[r], mx);
                float c  = __expf(mi[r] - mn);
                float es = 0.0f;
#pragma unroll
                for (int nt = 0; nt < 4; ++nt) {
                    float e = __expf(vals[nt] - mn);
                    pv[nt][r] = e;
                    es += e;
                }
                float sm = rowsum16(es);
                li[r] = li[r] * c + sm;
                mi[r] = mn;
                csc[r] = c;
            }
#pragma unroll
            for (int nt2 = 0; nt2 < 4; ++nt2)
#pragma unroll
                for (int r = 0; r < 8; ++r) o[nt2][r] *= csc[r];

            // ---- P: C-layout -> A-layout via per-wave LDS tile (16 x 64)
#pragma unroll
            for (int nt = 0; nt < 4; ++nt)
#pragma unroll
                for (int r = 0; r < 8; ++r)
                    Pw[(r + half * 8) * PSTR + nt * 16 + lr] = (__bf16)pv[nt][r];
            Frag16 pf[2];
#pragma unroll
            for (int ks = 0; ks < 2; ++ks) {
                const __bf16* pp = Pw + lr * PSTR + ks * 32 + half * 8;
                pf[ks].h[0] = *(const v8bf*)(pp);
                pf[ks].h[1] = *(const v8bf*)(pp + 16);
            }
            // ---- O += P V  (4 dh tiles x 2 k-steps): 8 WMMAs
#pragma unroll
            for (int nt2 = 0; nt2 < 4; ++nt2)
#pragma unroll
                for (int ks = 0; ks < 2; ++ks) {
                    const __bf16* vp = &Vlds[buf][0] + (nt2 * 16 + lr) * VSTR + ks * 32 + half * 8;
                    Frag16 b;
                    b.h[0] = *(const v8bf*)(vp);
                    b.h[1] = *(const v8bf*)(vp + 16);
                    o[nt2] = wmma_bf16(pf[ks].v, b.v, o[nt2]);
                }
        }
        __syncthreads();                  // reads done before buffer reuse
    }

    // ---- epilogue: bf16 [S][D] (feeds output-projection GEMM directly)
#pragma unroll
    for (int nt2 = 0; nt2 < 4; ++nt2)
#pragma unroll
        for (int r = 0; r < 8; ++r) {
            int row = qbase + r + half * 8;
            int col = colbase + nt2 * 16 + lr;
            aout[(size_t)row * D + col] = (__bf16)(o[nt2][r] / li[r]);
        }
}

// ---------------------------------------------------------------------------
// Host launcher
// ---------------------------------------------------------------------------
extern "C" void kernel_launch(void* const* d_in, const int* in_sizes, int n_in,
                              void* d_out, int out_size, void* d_ws, size_t ws_size,
                              hipStream_t stream) {
    const float* x     = (const float*)d_in[0];
    const float* rfreq = (const float*)d_in[1];
    const float* W_q   = (const float*)d_in[2];
    const float* W_k   = (const float*)d_in[3];
    const float* W_v   = (const float*)d_in[4];
    const float* W_o   = (const float*)d_in[5];
    float* out = (float*)d_out;

    char* w = (char*)d_ws;
    auto carve = [&](size_t bytes) {
        char* p = w;
        w += (bytes + 255) & ~(size_t)255;
        return p;
    };
    __bf16* wpack = (__bf16*)carve((size_t)4 * KDIM * NDIM * sizeof(__bf16));
    __bf16* xb    = (__bf16*)carve((size_t)S * D * sizeof(__bf16));
    float*  qraw  = (float*) carve((size_t)S * D * sizeof(float));
    float*  kraw  = (float*) carve((size_t)S * D * sizeof(float));
    float*  vraw  = (float*) carve((size_t)S * D * sizeof(float));
    __bf16* Qb    = (__bf16*)carve((size_t)S * D * sizeof(__bf16));
    __bf16* Kb    = (__bf16*)carve((size_t)S * D * sizeof(__bf16));
    __bf16* Vt    = (__bf16*)carve((size_t)S * D * sizeof(__bf16));
    __bf16* aoutb = (__bf16*)carve((size_t)S * D * sizeof(__bf16));

    // 1) pack weights into WMMA B fragments; convert x to bf16
    {
        int total = 4 * (NDIM / 16) * NKSTEPS * 32;
        pack_weights<<<(total + 255) / 256, 256, 0, stream>>>(W_q, W_k, W_v, W_o, wpack);
        int n4 = S * D / 4;
        convert_bf16<<<(n4 + 255) / 256, 256, 0, stream>>>(x, xb, n4);
    }
    // 2) Q/K/V projections
    dim3 ggrid(S / 128, NDIM / 128);
    gemm_bf16<<<ggrid, 128, 0, stream>>>(xb, wpack + (size_t)0 * KDIM * NDIM, qraw, S);
    gemm_bf16<<<ggrid, 128, 0, stream>>>(xb, wpack + (size_t)1 * KDIM * NDIM, kraw, S);
    gemm_bf16<<<ggrid, 128, 0, stream>>>(xb, wpack + (size_t)2 * KDIM * NDIM, vraw, S);
    // 3) RoPE + bf16 conversion (V transposed)
    {
        int total = S * H * (DH / 2);
        rope_pack<<<(total + 255) / 256, 256, 0, stream>>>(qraw, kraw, vraw, rfreq, Qb, Kb, Vt);
    }
    // 4) causal flash attention
    attention<<<dim3(S / 128, H), 256, 0, stream>>>(Qb, Kb, Vt, aoutb);
    // 5) output projection -> d_out (fp32)
    gemm_bf16<<<ggrid, 128, 0, stream>>>(aoutb, wpack + (size_t)3 * KDIM * NDIM, out, S);
}
